// GPTQLinear_833223655717
// MI455X (gfx1250) — matrix operand order repaired and checked by reading, not compiled, not run on hardware
//
#include <hip/hip_runtime.h>
#include <hip/hip_bf16.h>

typedef __attribute__((ext_vector_type(16))) _Float16 v16h;
typedef __attribute__((ext_vector_type(8)))  _Float16 v8h;
typedef __attribute__((ext_vector_type(8)))  float    v8f;
typedef __attribute__((ext_vector_type(4)))  int      v4i;

// Problem constants (fixed by the reference setup)
constexpr int M          = 32;
constexpr int K          = 8192;
constexpr int N          = 8192;
constexpr int GROUP_SIZE = 64;
constexpr int NUM_GROUPS = K / GROUP_SIZE;   // 128
constexpr int KSPLIT     = 8;                // K-chunks accumulated via f32 atomics
constexpr int KCHUNK     = K / KSPLIT;       // 1024 -> 32 iterations of 32-K

// ---------------------------------------------------------------------------
// Kernel 0: convert activations f32 -> f16 into workspace (x is tiny: 1 MB,
// stays L2-resident; f16 A feeds v_wmma_f32_16x16x32_f16 directly).
// ---------------------------------------------------------------------------
__global__ void __launch_bounds__(256) cvt_x_f16(const float* __restrict__ x,
                                                 _Float16* __restrict__ xh) {
    int i = blockIdx.x * 256 + threadIdx.x;   // M*K / 256 blocks, exact
    xh[i] = (_Float16)x[i];
}

// ---------------------------------------------------------------------------
// Kernel 1: out[m, n] = bias[n]  (K-split partials are atomically added later)
// ---------------------------------------------------------------------------
__global__ void __launch_bounds__(256) init_out(const float* __restrict__ bias,
                                                float* __restrict__ out) {
    int i = blockIdx.x * 256 + threadIdx.x;   // M*N / 256 blocks, exact
    out[i] = bias[i & (N - 1)];
}

// ---------------------------------------------------------------------------
// Kernel 2: main GPTQ GEMM.
//   block = 256 threads = 8 waves. Wave w owns N columns [blockIdx.x*128 + w*16, +16)
//   and both 16-row M tiles. grid = (N/128, KSPLIT).
//   Per 32-K iteration: dequant one 32x16 f16 B tile (int4 -> f16), two WMMAs.
// ---------------------------------------------------------------------------
__global__ void __launch_bounds__(256)
gptq_wmma(const _Float16* __restrict__ xh,     // [M, K] f16 (workspace)
          const int* __restrict__ pw,          // [N, K/2] one packed byte per int32
          const float* __restrict__ scales,    // [N, NUM_GROUPS]
          const float* __restrict__ zeros,     // [N, NUM_GROUPS]
          float* __restrict__ out) {           // [M, N] f32 (pre-seeded with bias)
    const int lane  = threadIdx.x & 31;
    const int wave  = threadIdx.x >> 5;
    const int lcol  = lane & 15;               // N within the wave's 16-col slice
    const int khalf = lane >> 4;               // 0: K {0..7,16..23}, 1: K {8..15,24..31}
    const int col   = blockIdx.x * 128 + wave * 16 + lcol;
    const int kbase = blockIdx.y * KCHUNK;

    // A-matrix (16x32 f16, §7.12.2): lane holds row (lane&15) of the tile,
    // two contiguous 8-half chunks at K = khalf*8 + {0..7} and {16..23}.
    const _Float16* a0p = xh + (size_t)lcol * K        + kbase + khalf * 8; // M tile 0
    const _Float16* a1p = xh + (size_t)(lcol + 16) * K + kbase + khalf * 8; // M tile 1

    // B-matrix (32x16 f16): lane holds column lcol, K = khalf*8 + {0..7} and {16..23}.
    // 16 K values = 8 packed bytes = int32 at bq[0..3] and bq[8..11].
    const int* bq = pw + (size_t)col * (K / 2) + (kbase + khalf * 8) / 2;

    const float* sp = scales + (size_t)col * NUM_GROUPS;
    const float* zp = zeros  + (size_t)col * NUM_GROUPS;

    v8f c0 = {};   // D for M rows 0..15
    v8f c1 = {};   // D for M rows 16..31

    for (int kk = 0; kk < KCHUNK; kk += 32) {
        // Keep the weight stream ahead of the ALUs (global_prefetch_b8).
        __builtin_prefetch(bq + 32, 0, 0);

        // ---- load + build A fragments (two B128 loads per tile) ----
        v8h a0lo = *(const v8h*)(a0p + kk);
        v8h a0hi = *(const v8h*)(a0p + kk + 16);
        v8h a1lo = *(const v8h*)(a1p + kk);
        v8h a1hi = *(const v8h*)(a1p + kk + 16);
        v16h A0 = __builtin_shufflevector(a0lo, a0hi,
                    0,1,2,3,4,5,6,7,8,9,10,11,12,13,14,15);
        v16h A1 = __builtin_shufflevector(a1lo, a1hi,
                    0,1,2,3,4,5,6,7,8,9,10,11,12,13,14,15);

        // ---- load packed weights: 8 int32 = 16 nibbles ----
        v4i b0 = *(const v4i*)bq;
        v4i b1 = *(const v4i*)(bq + 8);
        bq += 16;

        // Quant group for this lane's 16-K span (spans never cross a group).
        const int g = (kbase + kk + khalf * 16) >> 6;   // /GROUP_SIZE
        const float s  = sp[g];
        const float zs = zp[g] * s;                     // w = q*s - z*s

        // ---- dequant int4 -> f16 B fragment ----
        v16h B;
#pragma unroll
        for (int j = 0; j < 4; ++j) {
            int by = b0[j] & 0xFF;                      // even k: low nibble
            B[2 * j]     = (_Float16)fmaf((float)(by & 15),        s, -zs);
            B[2 * j + 1] = (_Float16)fmaf((float)((by >> 4) & 15), s, -zs);
        }
#pragma unroll
        for (int j = 0; j < 4; ++j) {
            int by = b1[j] & 0xFF;
            B[8 + 2 * j]     = (_Float16)fmaf((float)(by & 15),        s, -zs);
            B[8 + 2 * j + 1] = (_Float16)fmaf((float)((by >> 4) & 15), s, -zs);
        }

        // ---- two WMMAs share one dequantized B tile ----
        c0 = __builtin_amdgcn_wmma_f32_16x16x32_f16(
                 false, A0, false, B, (short)0, c0, false, false);
        c1 = __builtin_amdgcn_wmma_f32_16x16x32_f16(
                 false, A1, false, B, (short)0, c1, false, false);
    }

    // D layout (§7.12.2): VGPR r, lane l -> element (M = r + 8*(l>>4), N = l&15).
    const int orow = khalf * 8;
    const int ocol = blockIdx.x * 128 + wave * 16 + lcol;
#pragma unroll
    for (int r = 0; r < 8; ++r) {
        unsafeAtomicAdd(&out[(size_t)(orow + r) * N + ocol],      c0[r]);
        unsafeAtomicAdd(&out[(size_t)(orow + r + 16) * N + ocol], c1[r]);
    }
}

// ---------------------------------------------------------------------------
// Host launcher. Inputs (reference order): x, packed_weight, scales, zeros,
// bias_param. Output: [M, N] f32. Workspace: x as f16 (512 KB).
// ---------------------------------------------------------------------------
extern "C" void kernel_launch(void* const* d_in, const int* in_sizes, int n_in,
                              void* d_out, int out_size, void* d_ws, size_t ws_size,
                              hipStream_t stream) {
    const float* x      = (const float*)d_in[0];
    const int*   pw     = (const int*)d_in[1];
    const float* scales = (const float*)d_in[2];
    const float* zeros  = (const float*)d_in[3];
    const float* bias   = (const float*)d_in[4];
    float*       out    = (float*)d_out;
    _Float16*    xh     = (_Float16*)d_ws;

    cvt_x_f16<<<(M * K) / 256, 256, 0, stream>>>(x, xh);
    init_out <<<(M * N) / 256, 256, 0, stream>>>(bias, out);

    dim3 grid(N / 128, KSPLIT);
    gptq_wmma<<<grid, 256, 0, stream>>>(xh, pw, scales, zeros, out);
}
